// TripConstructionDecoder_20237885899395
// MI455X (gfx1250) — matrix operand order repaired and checked by reading, not compiled, not run on hardware
//
#include <hip/hip_runtime.h>

// ---------------------------------------------------------------------------
// TripConstructionDecoder for MI455X (gfx1250, wave32, WMMA bf16 16x16x32).
//
// Math restructuring (see analysis): fold wk / W_key into the query side so
// both heavy node-side GEMMs collapse into thin per-batch GEMM/GEMV passes
// over node_embeddings. Total HBM traffic ~2 passes over E (~2.1 GB).
// ---------------------------------------------------------------------------

#define BATCH 1024
#define NNODE 500
#define DIM   512
#define NH    8

typedef __attribute__((ext_vector_type(16))) __bf16 v16bf;
typedef __attribute__((ext_vector_type(8)))  float  v8f;

union Frag { v16bf v; uint4 u[2]; };

__device__ __forceinline__ unsigned short f2bf(float f) {
  unsigned int u = __float_as_uint(f);
  u += 0x7fffu + ((u >> 16) & 1u);          // round-to-nearest-even
  return (unsigned short)(u >> 16);
}
__device__ __forceinline__ float bf2f(unsigned short h) {
  return __uint_as_float(((unsigned int)h) << 16);
}
__device__ __forceinline__ float wave_max(float v) {
  #pragma unroll
  for (int d = 16; d; d >>= 1) v = fmaxf(v, __shfl_xor(v, d, 32));
  return v;
}
__device__ __forceinline__ float wave_sum(float v) {
  #pragma unroll
  for (int d = 16; d; d >>= 1) v += __shfl_xor(v, d, 32);
  return v;
}

// ---------------------------------------------------------------------------
// Weight conversion helpers (f32 -> bf16, optional zero row padding / slicing)
// ---------------------------------------------------------------------------
__global__ void cvt_copy(unsigned short* __restrict__ dst,
                         const float* __restrict__ src,
                         int dst_rows, int cols, int src_rows,
                         int src_ld, int src_col_off) {
  int i = blockIdx.x * blockDim.x + threadIdx.x;
  if (i >= dst_rows * cols) return;
  int r = i / cols, c = i - r * cols;
  float v = (r < src_rows) ? src[(size_t)r * src_ld + src_col_off + c] : 0.f;
  dst[i] = f2bf(v);
}

// dst[o][e] = src[e][src_col_off + o]   (builds W^T slices in bf16)
__global__ void cvt_transpose(unsigned short* __restrict__ dst,
                              const float* __restrict__ src,
                              int rows_o, int cols_e, int src_ld, int src_col_off) {
  int i = blockIdx.x * blockDim.x + threadIdx.x;
  if (i >= rows_o * cols_e) return;
  int o = i / cols_e, e = i - o * cols_e;
  dst[i] = f2bf(src[(size_t)e * src_ld + src_col_off + o]);
}

// Ctx[b][0:512]=graph, [512:1024]=last, [1024]=cap, [1025]=hours, pad->1056
__global__ __launch_bounds__(256) void build_ctx(
    const float* __restrict__ g, const float* __restrict__ l,
    const float* __restrict__ cap, const float* __restrict__ hrs,
    unsigned short* __restrict__ ctx) {
  int b = blockIdx.x;
  for (int c = threadIdx.x; c < 1056; c += 256) {
    float v;
    if (c < 512)       v = g[(size_t)b * 512 + c];
    else if (c < 1024) v = l[(size_t)b * 512 + c - 512];
    else if (c == 1024) v = cap[b];
    else if (c == 1025) v = hrs[b];
    else v = 0.f;
    ctx[(size_t)b * 1056 + c] = f2bf(v);
  }
}

// c[b][h] = bk_h . q_scaled[b,h,:]
__global__ __launch_bounds__(256) void chead_kernel(
    const unsigned short* __restrict__ qs, const float* __restrict__ bk,
    float* __restrict__ c) {
  int b = blockIdx.x, lane = threadIdx.x & 31, h = threadIdx.x >> 5;
  int d0 = lane * 2;
  float s = bf2f(qs[(size_t)b * 512 + h * 64 + d0])     * bk[h * 64 + d0]
          + bf2f(qs[(size_t)b * 512 + h * 64 + d0 + 1]) * bk[h * 64 + d0 + 1];
  s = wave_sum(s);
  if (lane == 0) c[b * NH + h] = s;
}

// ---------------------------------------------------------------------------
// Generic tiled WMMA bf16 GEMM:  C[z] = (A[z] @ B[z] + bias[z]) * scale
//   BM=128, BN=64, BK=32; 256 threads = 8 waves; wave w owns rows w*16.
//   A global row-major [M x (lda)], B global row-major [K x (ldb)].
//   B is staged into LDS transposed (N-major) so fragments are contiguous.
//   Per-z offsets express per-head block-diagonal contractions.
// ---------------------------------------------------------------------------
template <bool OUT_BF16>
__global__ __launch_bounds__(256) void gemm_bf16(
    const unsigned short* __restrict__ A, int lda, int aColOffZ,
    const unsigned short* __restrict__ Bm, int ldb, int bRowOffZ, int bColOffZ,
    void* __restrict__ Cp, int ldc, int cColOffZ,
    const float* __restrict__ bias, int biasOffZ, float scale, int K) {
  __shared__ unsigned short sA[128 * 40];   // rows padded to 40 (16B aligned)
  __shared__ unsigned short sB[64 * 40];    // [n][k] transposed-stored
  const int tid = threadIdx.x, lane = tid & 31, wv = tid >> 5;
  const int lo = lane & 15, hi = lane >> 4;
  const int z = blockIdx.z;
  const int m0 = blockIdx.y * 128, n0 = blockIdx.x * 64;
  const int aOff = aColOffZ * z, bR = bRowOffZ * z, bC = bColOffZ * z;

  v8f acc[4] = {};

  for (int k0 = 0; k0 < K; k0 += 32) {
    __syncthreads();
    {   // stage A tile: 128x32 bf16
      int row = tid >> 1, half = tid & 1;
      const unsigned short* src =
          A + (size_t)(m0 + row) * lda + aOff + k0 + half * 16;
      uint4 u0 = *(const uint4*)src;
      uint4 u1 = *(const uint4*)(src + 8);
      *(uint4*)(sA + row * 40 + half * 16) = u0;
      *(uint4*)(sA + row * 40 + half * 16 + 8) = u1;
    }
    {   // stage B tile transposed: [n][k]
      int kk = tid & 31, nb = (tid >> 5) * 8;
      const unsigned short* src =
          Bm + (size_t)(k0 + kk + bR) * ldb + n0 + bC + nb;
      unsigned short tmp[8];
      *(uint4*)tmp = *(const uint4*)src;
      #pragma unroll
      for (int q = 0; q < 8; ++q) sB[(nb + q) * 40 + kk] = tmp[q];
    }
    __syncthreads();

    Frag Af;   // 16-bit A 16x32: lane row = lo, K halves selected by hi
    int abase = (wv * 16 + lo) * 40 + hi * 8;
    Af.u[0] = *(const uint4*)(sA + abase);
    Af.u[1] = *(const uint4*)(sA + abase + 16);
    #pragma unroll
    for (int i = 0; i < 4; ++i) {
      Frag Bf; // 16-bit B 32x16: lane col = lo, K halves selected by hi
      int bbase = (i * 16 + lo) * 40 + hi * 8;
      Bf.u[0] = *(const uint4*)(sB + bbase);
      Bf.u[1] = *(const uint4*)(sB + bbase + 16);
      acc[i] = __builtin_amdgcn_wmma_f32_16x16x32_bf16(
          false, Af.v, false, Bf.v, (short)0, acc[i], false, false);
    }
  }

  // C layout: VGPR j -> row j (lanes 0-15) / row 8+j (lanes 16-31), col = lo
  #pragma unroll
  for (int i = 0; i < 4; ++i) {
    int col = n0 + i * 16 + lo;
    float bv = bias ? bias[biasOffZ * z + col] : 0.f;
    #pragma unroll
    for (int j = 0; j < 8; ++j) {
      int row = m0 + wv * 16 + j + hi * 8;
      float v = (acc[i][j] + bv) * scale;
      size_t idx = (size_t)row * ldc + cColOffZ * z + col;
      if (OUT_BF16) ((unsigned short*)Cp)[idx] = f2bf(v);
      else          ((float*)Cp)[idx] = v;
    }
  }
}

// ---------------------------------------------------------------------------
// Fused per-batch single-query attention (pass 1 over E):
//   scores = E_tile @ P^T (+c_h), online softmax per head across node tiles,
//   S_h += attn_tile @ E_tile  (WMMA, heads along M), output sbar = S_h / Z_h.
// One block (8 waves) per batch; wave w tracks head w's running max/sum.
// ---------------------------------------------------------------------------
#define EB_LD 520   // 64 node rows, padded (16B aligned, conflict-free)
#define ET_LD 80    // E^T rows (e), 64 node cols padded
#define PB_LD 520
#define AT_LD 80
#define ATTN_SMEM ((64*EB_LD + 512*ET_LD + 16*PB_LD + 16*AT_LD)*2 + 64*17*4 + 64)

__global__ __launch_bounds__(256) void attn_kernel(
    const float* __restrict__ E,              // [B,N,D] f32
    const unsigned short* __restrict__ Pw,    // [B,8,512] bf16
    const float* __restrict__ chead,          // [B,8]
    const int* __restrict__ mask,             // [B,N] (nonzero = infeasible)
    unsigned short* __restrict__ sbar) {      // [B,8,512] bf16
  extern __shared__ char smem[];
  unsigned short* Ebuf  = (unsigned short*)smem;        // [64][EB_LD] row-major
  unsigned short* Et    = Ebuf + 64 * EB_LD;            // [512][ET_LD] E^T
  unsigned short* Pb    = Et + 512 * ET_LD;             // [16][PB_LD] (P, B-frag form)
  unsigned short* attnA = Pb + 16 * PB_LD;              // [16][AT_LD] softmax weights
  float* scor    = (float*)(attnA + 16 * AT_LD);        // [64][17]
  float* alpha_l = scor + 64 * 17;                      // [8]
  float* zinv_l  = alpha_l + 8;                         // [8]

  const int b = blockIdx.x;
  const int tid = threadIdx.x, lane = tid & 31, wv = tid >> 5;
  const int lo = lane & 15, hi = lane >> 4;

  // Stage P[b] (8x512) into B-fragment form; pad heads 8..15 with zeros.
  for (int i = tid; i < 16 * 512; i += 256) {
    int r = i >> 9, c = i & 511;
    Pb[r * PB_LD + c] = (r < 8) ? Pw[((size_t)b * 8 + r) * 512 + c]
                                : (unsigned short)0;
  }
  for (int i = tid; i < 16 * AT_LD; i += 256) attnA[i] = 0;

  v8f accS[4] = {};                 // per-wave: 4 e-subtiles of S (16x16 f32)
  float m_run = -__builtin_inff();  // wave wv <-> head wv
  float z_run = 0.f;
  const float ch = chead[b * NH + wv];

  const int NT = (NNODE + 63) / 64;
  for (int t = 0; t < NT; ++t) {
    const int n0 = t * 64;
    __syncthreads();
    // ---- stage E tile (64x512 f32 -> bf16, row-major + transposed) ----
    for (int i = tid; i < (64 * 512) / 4; i += 256) {
      int lin = i * 4;
      int node = lin >> 9, e = lin & 511;
      int gn = n0 + node;
      float4 val;
      if (gn < NNODE) {
        val = *(const float4*)(E + (((size_t)b * NNODE + gn) << 9) + e);
        if ((e & 31) == 0 && gn + 64 < NNODE)  // prefetch next tile
          __builtin_prefetch(E + (((size_t)b * NNODE + gn + 64) << 9) + e, 0, 0);
      } else {
        val = make_float4(0.f, 0.f, 0.f, 0.f);
      }
      unsigned short h0 = f2bf(val.x), h1 = f2bf(val.y),
                     h2 = f2bf(val.z), h3 = f2bf(val.w);
      unsigned long long pk = (unsigned long long)h0 |
                              ((unsigned long long)h1 << 16) |
                              ((unsigned long long)h2 << 32) |
                              ((unsigned long long)h3 << 48);
      *(unsigned long long*)(Ebuf + node * EB_LD + e) = pk;
      Et[(e + 0) * ET_LD + node] = h0;
      Et[(e + 1) * ET_LD + node] = h1;
      Et[(e + 2) * ET_LD + node] = h2;
      Et[(e + 3) * ET_LD + node] = h3;
    }
    __syncthreads();

    // ---- scores: [64 nodes x 16 (8 heads)] = E_tile @ P^T, waves 0-3 ----
    if (wv < 4) {
      v8f acc = {};
      const int row = wv * 16 + lo;
      for (int ks = 0; ks < 16; ++ks) {
        Frag Af;
        int abase = row * EB_LD + ks * 32 + hi * 8;
        Af.u[0] = *(const uint4*)(Ebuf + abase);
        Af.u[1] = *(const uint4*)(Ebuf + abase + 16);
        Frag Bf;
        int bbase = lo * PB_LD + ks * 32 + hi * 8;
        Bf.u[0] = *(const uint4*)(Pb + bbase);
        Bf.u[1] = *(const uint4*)(Pb + bbase + 16);
        acc = __builtin_amdgcn_wmma_f32_16x16x32_bf16(
            false, Af.v, false, Bf.v, (short)0, acc, false, false);
      }
      #pragma unroll
      for (int j = 0; j < 8; ++j) {
        int node = wv * 16 + j + hi * 8;
        scor[node * 17 + lo] = acc[j];
      }
    }
    __syncthreads();

    // ---- online softmax: wave wv = head wv, 2 nodes per lane ----
    {
      int nA = 2 * lane, nB = 2 * lane + 1;
      int gA = n0 + nA, gB = n0 + nB;
      bool vA = (gA < NNODE) && (mask[(size_t)b * NNODE + gA] == 0);
      bool vB = (gB < NNODE) && (mask[(size_t)b * NNODE + gB] == 0);
      float s0 = vA ? scor[nA * 17 + wv] + ch : -__builtin_inff();
      float s1 = vB ? scor[nB * 17 + wv] + ch : -__builtin_inff();
      float mx = wave_max(fmaxf(s0, s1));
      float newm = fmaxf(m_run, mx);
      float al, w0, w1;
      if (newm == -__builtin_inff()) { al = 1.f; w0 = 0.f; w1 = 0.f; }
      else {
        al = __expf(m_run - newm);
        w0 = vA ? __expf(s0 - newm) : 0.f;
        w1 = vB ? __expf(s1 - newm) : 0.f;
      }
      float zs = wave_sum(w0 + w1);
      m_run = newm;
      z_run = z_run * al + zs;
      attnA[wv * AT_LD + nA] = f2bf(w0);
      attnA[wv * AT_LD + nB] = f2bf(w1);
      if (lane == 0) alpha_l[wv] = al;
    }
    __syncthreads();

    // ---- S update: S = S*alpha_row + attn @ E_tile (all 8 waves) ----
    float a8[8];
    #pragma unroll
    for (int j = 0; j < 8; ++j) a8[j] = alpha_l[j];
    #pragma unroll
    for (int i = 0; i < 4; ++i)
      #pragma unroll
      for (int j = 0; j < 8; ++j) accS[i][j] *= a8[j];  // row j == head j

    #pragma unroll
    for (int ks = 0; ks < 2; ++ks) {
      Frag Af;
      int abase = lo * AT_LD + ks * 32 + hi * 8;
      Af.u[0] = *(const uint4*)(attnA + abase);
      Af.u[1] = *(const uint4*)(attnA + abase + 16);
      #pragma unroll
      for (int i = 0; i < 4; ++i) {
        int ecol = (wv * 4 + i) * 16 + lo;
        Frag Bf;
        int bbase = ecol * ET_LD + ks * 32 + hi * 8;
        Bf.u[0] = *(const uint4*)(Et + bbase);
        Bf.u[1] = *(const uint4*)(Et + bbase + 16);
        accS[i] = __builtin_amdgcn_wmma_f32_16x16x32_bf16(
            false, Af.v, false, Bf.v, (short)0, accS[i], false, false);
      }
    }
  }

  // ---- finalize: sbar = S / Z ----
  if (lane == 0) zinv_l[wv] = (z_run > 0.f) ? 1.f / z_run : 0.f;
  __syncthreads();
  float zi[8];
  #pragma unroll
  for (int j = 0; j < 8; ++j) zi[j] = zinv_l[j];
  if (hi == 0) {  // lanes 0-15 hold heads 0-7 (rows 8-15 are padding)
    #pragma unroll
    for (int i = 0; i < 4; ++i) {
      int ecol = (wv * 4 + i) * 16 + lo;
      #pragma unroll
      for (int j = 0; j < 8; ++j)
        sbar[((size_t)b * 8 + j) * 512 + ecol] = f2bf(accS[i][j] * zi[j]);
    }
  }
}

// ---------------------------------------------------------------------------
// Pass 2 over E: logits[b,n] = masked(10*tanh((E[b,n,:].r[b]) / sqrt(512)))
// One block per batch; wave per node; lane covers 16 contiguous e.
// ---------------------------------------------------------------------------
__global__ __launch_bounds__(256) void logits_kernel(
    const float* __restrict__ E, const float* __restrict__ R,
    const int* __restrict__ mask, float* __restrict__ out) {
  __shared__ float rs[512];
  const int b = blockIdx.x, tid = threadIdx.x, lane = tid & 31, wv = tid >> 5;
  for (int i = tid; i < 512; i += 256) rs[i] = R[(size_t)b * 512 + i];
  __syncthreads();
  const float invs = 0.04419417382415922f;  // 1/sqrt(512)
  for (int n = wv; n < NNODE; n += 8) {
    const float* row = E + ((size_t)b * NNODE + n) * 512 + lane * 16;
    float s = 0.f;
    #pragma unroll
    for (int q = 0; q < 4; ++q) {
      float4 v = ((const float4*)row)[q];
      const float* r4 = rs + lane * 16 + q * 4;
      s += v.x * r4[0] + v.y * r4[1] + v.z * r4[2] + v.w * r4[3];
    }
    s = wave_sum(s);
    if (lane == 0) {
      float l = 10.f * tanhf(s * invs);
      out[(size_t)b * NNODE + n] =
          mask[(size_t)b * NNODE + n] ? -__builtin_inff() : l;
    }
  }
}

// ---------------------------------------------------------------------------
extern "C" void kernel_launch(void* const* d_in, const int* in_sizes, int n_in,
                              void* d_out, int out_size, void* d_ws,
                              size_t ws_size, hipStream_t stream) {
  (void)in_sizes; (void)n_in; (void)out_size; (void)ws_size;
  const float* E    = (const float*)d_in[0];
  const float* Gemb = (const float*)d_in[1];
  const float* Lemb = (const float*)d_in[2];
  const float* Cap  = (const float*)d_in[3];
  const float* Hrs  = (const float*)d_in[4];
  const int*   Mask = (const int*)d_in[5];
  const float* Wctx = (const float*)d_in[6];
  const float* Win  = (const float*)d_in[7];   // [512][1536] packed q|k|v
  const float* Bin  = (const float*)d_in[8];   // [1536]
  const float* Wout = (const float*)d_in[9];
  const float* Bout = (const float*)d_in[10];
  const float* Wqry = (const float*)d_in[11];
  const float* Wkey = (const float*)d_in[12];
  float* out = (float*)d_out;

  char* p = (char*)d_ws;
  auto take = [&](size_t bytes) {
    char* r = p; p += (bytes + 255) & ~(size_t)255; return r;
  };
  unsigned short* WC    = (unsigned short*)take((size_t)1056 * 512 * 2);
  unsigned short* WQ    = (unsigned short*)take((size_t)512 * 512 * 2);
  unsigned short* WKT   = (unsigned short*)take((size_t)512 * 512 * 2);
  unsigned short* WV    = (unsigned short*)take((size_t)512 * 512 * 2);
  unsigned short* WO    = (unsigned short*)take((size_t)512 * 512 * 2);
  unsigned short* WQU   = (unsigned short*)take((size_t)512 * 512 * 2);
  unsigned short* WKEYT = (unsigned short*)take((size_t)512 * 512 * 2);
  unsigned short* CTX   = (unsigned short*)take((size_t)1024 * 1056 * 2);
  unsigned short* Q0    = (unsigned short*)take((size_t)1024 * 512 * 2);
  unsigned short* QS    = (unsigned short*)take((size_t)1024 * 512 * 2);
  float*          CH    = (float*)take((size_t)1024 * 8 * 4);
  unsigned short* PW    = (unsigned short*)take((size_t)1024 * 4096 * 2);
  unsigned short* SBAR  = (unsigned short*)take((size_t)1024 * 4096 * 2);
  unsigned short* GLI   = (unsigned short*)take((size_t)1024 * 512 * 2);
  unsigned short* G2    = (unsigned short*)take((size_t)1024 * 512 * 2);
  unsigned short* QG    = (unsigned short*)take((size_t)1024 * 512 * 2);
  float*          RV    = (float*)take((size_t)1024 * 512 * 4);

  dim3 blk(256);
  auto g1 = [](int total) { return dim3((total + 255) / 256); };

  // weights -> bf16 (WKT = wk^T, WKEYT = W_key^T for query-side folding)
  cvt_copy<<<g1(1056 * 512), blk, 0, stream>>>(WC, Wctx, 1056, 512, 1026, 512, 0);
  cvt_copy<<<g1(512 * 512), blk, 0, stream>>>(WQ, Win, 512, 512, 512, 1536, 0);
  cvt_copy<<<g1(512 * 512), blk, 0, stream>>>(WV, Win, 512, 512, 512, 1536, 1024);
  cvt_copy<<<g1(512 * 512), blk, 0, stream>>>(WO, Wout, 512, 512, 512, 512, 0);
  cvt_copy<<<g1(512 * 512), blk, 0, stream>>>(WQU, Wqry, 512, 512, 512, 512, 0);
  cvt_transpose<<<g1(512 * 512), blk, 0, stream>>>(WKT, Win, 512, 512, 1536, 512);
  cvt_transpose<<<g1(512 * 512), blk, 0, stream>>>(WKEYT, Wkey, 512, 512, 512, 0);
  build_ctx<<<dim3(1024), blk, 0, stream>>>(Gemb, Lemb, Cap, Hrs, CTX);

  // q0 = ctx @ W_context                       [1024x1056]@[1056x512]
  gemm_bf16<true><<<dim3(8, 8, 1), blk, 0, stream>>>(
      CTX, 1056, 0, WC, 512, 0, 0, Q0, 512, 0, nullptr, 0, 1.f, 1056);
  // q = (q0 @ wq + bq) * DH^-0.5               [1024x512]@[512x512]
  gemm_bf16<true><<<dim3(8, 8, 1), blk, 0, stream>>>(
      Q0, 512, 0, WQ, 512, 0, 0, QS, 512, 0, Bin, 0, 0.125f, 512);
  // c_h = bk_h . q_h
  chead_kernel<<<dim3(1024), blk, 0, stream>>>(QS, Bin + 512, CH);
  // P_h = q_h @ wk_h^T  (per-head, z = head)   8x [1024x64]@[64x512]
  gemm_bf16<true><<<dim3(8, 8, 8), blk, 0, stream>>>(
      QS, 512, 64, WKT, 512, 64, 0, PW, 4096, 512, nullptr, 0, 1.f, 64);

  // fused attention pass over E -> sbar
  hipFuncSetAttribute((const void*)attn_kernel,
                      hipFuncAttributeMaxDynamicSharedMemorySize, ATTN_SMEM);
  attn_kernel<<<dim3(1024), blk, ATTN_SMEM, stream>>>(E, PW, CH, Mask, SBAR);

  // glimpse_h = sbar_h @ wv_h + bv_h           8x [1024x512]@[512x64]
  gemm_bf16<true><<<dim3(1, 8, 8), blk, 0, stream>>>(
      SBAR, 4096, 512, WV, 512, 0, 64, GLI, 512, 64, Bin + 1024, 64, 1.f, 512);
  // g2 = glimpse @ out_proj + b
  gemm_bf16<true><<<dim3(8, 8, 1), blk, 0, stream>>>(
      GLI, 512, 0, WO, 512, 0, 0, G2, 512, 0, Bout, 0, 1.f, 512);
  // qg = g2 @ W_query
  gemm_bf16<true><<<dim3(8, 8, 1), blk, 0, stream>>>(
      G2, 512, 0, WQU, 512, 0, 0, QG, 512, 0, nullptr, 0, 1.f, 512);
  // r = qg @ W_key^T  (f32 output)
  gemm_bf16<false><<<dim3(8, 8, 1), blk, 0, stream>>>(
      QG, 512, 0, WKEYT, 512, 0, 0, RV, 512, 0, nullptr, 0, 1.f, 512);

  // pass 2 over E -> logits
  logits_kernel<<<dim3(1024), blk, 0, stream>>>(E, RV, Mask, out);
}
